// ClientModel_14293651161162
// MI455X (gfx1250) — matrix operand order repaired and checked by reading, not compiled, not run on hardware
//
#include <hip/hip_runtime.h>

typedef __attribute__((ext_vector_type(16))) _Float16 v16h;
typedef __attribute__((ext_vector_type(8)))  _Float16 v8h;
typedef __attribute__((ext_vector_type(8)))  float    v8f;

#define VOCAB 400001
#define H     100
#define BSZ   512
#define T     25
#define FCN   128
#define NC    2

#define BT    32            // batch rows per workgroup
#define NWG   (BSZ / BT)    // 16 workgroups
#define KT_L  7             // K tiles for LSTM GEMM (224 / 32)
#define NT_L  25            // valid N tiles for LSTM GEMM (400 / 16)
#define NTP_L 26            // padded N tiles (even, for pairing; tile 25 = zeros)
#define KT_F  4             // K tiles for fc1 (128 / 32)
#define NT_F  8             // N tiles for fc1 (128 / 16)
#define ASTR  232           // A-buffer row stride (halves), 16B-aligned rows
#define ZSTR  404           // z-buffer row stride (floats)
#define NTHREADS 256
#define NWAVES   8

// packed-weight workspace layout (in _Float16 elements)
#define K1_OFF 0
#define K1_SZ  (KT_L * NTP_L * 512)
#define K2_OFF (K1_OFF + K1_SZ)
#define K2_SZ  (KT_L * NTP_L * 512)
#define WF_OFF (K2_OFF + K2_SZ)
#define WF_SZ  (KT_F * NT_F * 512)

// ---------------------------------------------------------------------------
// Prep: convert fp32 weights into f16 WMMA B-fragments.
// ISA 16-bit B layout (32x16 tile, wave32): N = lane&15, K = 8*(lane>>4)+(hh&7)+16*(hh>>3),
// 16 contiguous halves per lane per fragment -> one v16h (2x b128) load later.
// Packed index = ((nt*KT + kt)*32 + lane)*16 + hh. Out-of-range K/N -> 0.
// ---------------------------------------------------------------------------
__global__ void pack_wmma_b(const float* __restrict__ src, _Float16* __restrict__ dst,
                            int Klim, int Nvalid, int Ncols, int KT, int NTpad) {
  int idx = blockIdx.x * blockDim.x + threadIdx.x;
  int total = KT * NTpad * 512;
  if (idx >= total) return;
  int hh   = idx & 15;
  int lane = (idx >> 4) & 31;
  int frag = idx >> 9;
  int kt = frag % KT;
  int nt = frag / KT;
  int K = kt * 32 + ((lane >> 4) << 3) + (hh & 7) + ((hh >> 3) << 4);
  int N = nt * 16 + (lane & 15);
  float v = (K < Klim && N < Nvalid) ? src[K * Ncols + N] : 0.0f;
  dst[idx] = (_Float16)v;
}

__device__ __forceinline__ float sigf(float x) { return 1.0f / (1.0f + __expf(-x)); }

// Load one 16x32 f16 A-fragment from an LDS row-major buffer.
__device__ __forceinline__ v16h load_a_frag(const _Float16* Abuf, int rowbase, int kt, int lane) {
  int kb = kt * 32 + ((lane >> 4) << 3);
  v8h lo = *(const v8h*)&Abuf[rowbase + kb];        // K 0..7   (+8 for hi lanes)
  v8h hi = *(const v8h*)&Abuf[rowbase + kb + 16];   // K 16..23 (+8 for hi lanes)
  return __builtin_shufflevector(lo, hi, 0, 1, 2, 3, 4, 5, 6, 7,
                                         8, 9, 10, 11, 12, 13, 14, 15);
}

// ---------------------------------------------------------------------------
// Persistent per-batch-tile LSTM. One WG owns 32 batch rows for all 25 steps;
// c/h state lives in LDS, weights stream from L2 as pre-packed B fragments.
// Each wave computes PAIRS of adjacent N-tiles: the A fragment is loaded once
// per kt and feeds two independent WMMA accumulation chains (latency hiding,
// half the LDS traffic per WMMA).
// ---------------------------------------------------------------------------
__global__ void __launch_bounds__(NTHREADS, 1)
lstm_wmma_kernel(const int*   __restrict__ feat,
                 const float* __restrict__ emb,
                 const float* __restrict__ b1,
                 const float* __restrict__ b2,
                 const float* __restrict__ bfc1,
                 const float* __restrict__ wfc2,
                 const float* __restrict__ bfc2,
                 const _Float16* __restrict__ wpack,
                 float* __restrict__ out) {
  __shared__ _Float16 A1[BT * ASTR];   // [x_t | h1 | zero-pad] f16
  __shared__ _Float16 A2[BT * ASTR];   // [h1  | h2 | zero-pad] f16
  __shared__ float    Z [BT * ZSTR];   // pre-activation gates z (fp32)
  __shared__ float    C1[BT * H];
  __shared__ float    C2[BT * H];

  const int tid  = threadIdx.x;
  const int lane = tid & 31;
  const int wid  = tid >> 5;
  const int wg   = blockIdx.x;

  for (int e = tid; e < BT * ASTR; e += NTHREADS) {
    A1[e] = (_Float16)0.0f;
    A2[e] = (_Float16)0.0f;
  }
  for (int e = tid; e < BT * H; e += NTHREADS) { C1[e] = 0.0f; C2[e] = 0.0f; }
  __syncthreads();

  const _Float16* k1p = wpack + K1_OFF;
  const _Float16* k2p = wpack + K2_OFF;
  const _Float16* wfp = wpack + WF_OFF;

  for (int t = 0; t < T; ++t) {
    // Gather x_t = embedding[features[:, t]] into A1[:, 0:H) as f16.
    for (int e = tid; e < BT * H; e += NTHREADS) {
      int b = e / H, u = e - b * H;
      int row = feat[(wg * BT + b) * T + t];
      A1[b * ASTR + u] = (_Float16)emb[row * H + u];
    }
    __syncthreads();

    for (int layer = 0; layer < 2; ++layer) {
      const _Float16* Abuf = layer ? A2 : A1;
      const _Float16* kp   = layer ? k2p : k1p;
      const float*    bb   = layer ? b2 : b1;
      float*          Cb   = layer ? C2 : C1;

      // GEMM: Z[32,400] = Abuf[32,224] @ W[224,400]  (f16 in, fp32 acc)
      // 2 M-tiles x 13 N-tile-pairs = 26 pairs across 8 waves.
      for (int p = wid; p < 2 * (NTP_L / 2); p += NWAVES) {
        int mt  = p / (NTP_L / 2);
        int np  = p - mt * (NTP_L / 2);
        int nt0 = np * 2;
        int nt1 = nt0 + 1;
        v8f acc0 = {};
        v8f acc1 = {};
        int rowbase = (mt * 16 + (lane & 15)) * ASTR;
        for (int kt = 0; kt < KT_L; ++kt) {
          v16h a  = load_a_frag(Abuf, rowbase, kt, lane);
          v16h b0 = *(const v16h*)&kp[((nt0 * KT_L + kt) * 32 + lane) * 16];
          v16h bq = *(const v16h*)&kp[((nt1 * KT_L + kt) * 32 + lane) * 16];
          if (kt + 1 < KT_L) {
            __builtin_prefetch(&kp[((nt0 * KT_L + kt + 1) * 32 + lane) * 16], 0, 3);
            __builtin_prefetch(&kp[((nt1 * KT_L + kt + 1) * 32 + lane) * 16], 0, 3);
          }
          acc0 = __builtin_amdgcn_wmma_f32_16x16x32_f16(false, a, false, b0,
                                                        (short)0, acc0, false, false);
          acc1 = __builtin_amdgcn_wmma_f32_16x16x32_f16(false, a, false, bq,
                                                        (short)0, acc1, false, false);
        }
        int m0 = mt * 16 + ((lane >> 4) << 3);
        int n0 = nt0 * 16 + (lane & 15);
#pragma unroll
        for (int r = 0; r < 8; ++r) Z[(m0 + r) * ZSTR + n0] = acc0[r];
        if (nt1 < NT_L) {                 // wave-uniform guard for padded tile
          int n1 = nt1 * 16 + (lane & 15);
#pragma unroll
          for (int r = 0; r < 8; ++r) Z[(m0 + r) * ZSTR + n1] = acc1[r];
        }
      }
      __syncthreads();

      // Gate math: c' = c*sig(f+1) + sig(i)*tanh(j); h = tanh(c')*sig(o)
      for (int e = tid; e < BT * H; e += NTHREADS) {
        int b = e / H, u = e - b * H;
        float zi = Z[b * ZSTR + u]         + bb[u];
        float zj = Z[b * ZSTR + u + H]     + bb[u + H];
        float zf = Z[b * ZSTR + u + 2 * H] + bb[u + 2 * H];
        float zo = Z[b * ZSTR + u + 3 * H] + bb[u + 3 * H];
        float c  = Cb[e];
        float cn = c * sigf(zf + 1.0f) + sigf(zi) * tanhf(zj);
        float h  = tanhf(cn) * sigf(zo);
        Cb[e] = cn;
        _Float16 h16 = (_Float16)h;
        if (layer == 0) {
          A1[b * ASTR + H + u] = h16;   // recurrent input for layer1, next step
          A2[b * ASTR + u]     = h16;   // feed-forward input for layer2, this step
        } else {
          A2[b * ASTR + H + u] = h16;   // recurrent input for layer2, next step
        }
      }
      __syncthreads();
    }
  }

  // FC head. Copy h2 into A1[:, 0:128) (zero pad 100..127) for a K=128 WMMA GEMM.
  for (int e = tid; e < BT * FCN; e += NTHREADS) {
    int b = e / FCN, u = e - b * FCN;
    A1[b * ASTR + u] = (u < H) ? A2[b * ASTR + H + u] : (_Float16)0.0f;
  }
  __syncthreads();

  // fc1 = h2 @ w_fc1 + b_fc1 -> Z[32,128].  2 M-tiles x 4 N-pairs = 8 pairs.
  for (int p = wid; p < 2 * (NT_F / 2); p += NWAVES) {
    int mt  = p / (NT_F / 2);
    int np  = p - mt * (NT_F / 2);
    int nt0 = np * 2;
    int nt1 = nt0 + 1;
    v8f acc0 = {};
    v8f acc1 = {};
    int rowbase = (mt * 16 + (lane & 15)) * ASTR;
    for (int kt = 0; kt < KT_F; ++kt) {
      v16h a  = load_a_frag(A1, rowbase, kt, lane);
      v16h b0 = *(const v16h*)&wfp[((nt0 * KT_F + kt) * 32 + lane) * 16];
      v16h bq = *(const v16h*)&wfp[((nt1 * KT_F + kt) * 32 + lane) * 16];
      acc0 = __builtin_amdgcn_wmma_f32_16x16x32_f16(false, a, false, b0,
                                                    (short)0, acc0, false, false);
      acc1 = __builtin_amdgcn_wmma_f32_16x16x32_f16(false, a, false, bq,
                                                    (short)0, acc1, false, false);
    }
    int m0 = mt * 16 + ((lane >> 4) << 3);
    int n0 = nt0 * 16 + (lane & 15);
    int n1 = nt1 * 16 + (lane & 15);
    float bias0 = bfc1[n0];
    float bias1 = bfc1[n1];
#pragma unroll
    for (int r = 0; r < 8; ++r) {
      Z[(m0 + r) * ZSTR + n0] = acc0[r] + bias0;
      Z[(m0 + r) * ZSTR + n1] = acc1[r] + bias1;
    }
  }
  __syncthreads();

  // fc2: [32,128] @ [128,2] + b_fc2 -> logits
  if (tid < BT * NC) {
    int b = tid >> 1, c = tid & 1;
    float s = bfc2[c];
    for (int k = 0; k < FCN; ++k) s += Z[b * ZSTR + k] * wfc2[k * NC + c];
    out[(wg * BT + b) * NC + c] = s;
  }
}

extern "C" void kernel_launch(void* const* d_in, const int* in_sizes, int n_in,
                              void* d_out, int out_size, void* d_ws, size_t ws_size,
                              hipStream_t stream) {
  (void)in_sizes; (void)n_in; (void)out_size; (void)ws_size;
  const int*   feat = (const int*)  d_in[0];
  const float* emb  = (const float*)d_in[1];
  const float* k1   = (const float*)d_in[2];
  const float* b1   = (const float*)d_in[3];
  const float* k2   = (const float*)d_in[4];
  const float* b2   = (const float*)d_in[5];
  const float* wfc1 = (const float*)d_in[6];
  const float* bfc1 = (const float*)d_in[7];
  const float* wfc2 = (const float*)d_in[8];
  const float* bfc2 = (const float*)d_in[9];
  float*    out   = (float*)d_out;
  _Float16* wpack = (_Float16*)d_ws;

  pack_wmma_b<<<(K1_SZ + 255) / 256, 256, 0, stream>>>(k1, wpack + K1_OFF,
                                                       2 * H, 4 * H, 4 * H, KT_L, NTP_L);
  pack_wmma_b<<<(K2_SZ + 255) / 256, 256, 0, stream>>>(k2, wpack + K2_OFF,
                                                       2 * H, 4 * H, 4 * H, KT_L, NTP_L);
  pack_wmma_b<<<(WF_SZ + 255) / 256, 256, 0, stream>>>(wfc1, wpack + WF_OFF,
                                                       H, FCN, FCN, KT_F, NT_F);

  lstm_wmma_kernel<<<NWG, NTHREADS, 0, stream>>>(feat, emb, b1, b2, bfc1, wfc2, bfc2, wpack, out);
}
